// MultiHeadSelfAttentionWithBias_37709812858994
// MI455X (gfx1250) — matrix operand order repaired and checked by reading, not compiled, not run on hardware
//
#include <hip/hip_runtime.h>
#include <hip/hip_bf16.h>

// Problem constants (B=2, S=2048, E=1024, H=16, DH=64)
#define BB   2
#define SS   2048
#define EE   1024
#define HH   16
#define DH   64

typedef __attribute__((ext_vector_type(16))) __bf16       v16bf;
typedef __attribute__((ext_vector_type(8)))  float        v8f;
typedef __attribute__((ext_vector_type(4)))  unsigned int v4u;

union Frag {
    v16bf bf;
    v4u   u[2];   // 32 bytes = 16 bf16
};

__device__ __forceinline__ unsigned short f32_to_bf16_rne(float f) {
    unsigned int u = __float_as_uint(f);
    u += 0x7FFFu + ((u >> 16) & 1u);
    return (unsigned short)(u >> 16);
}

__device__ __forceinline__ v8f wmma_bf16(const Frag& a, const Frag& b, v8f c) {
    return __builtin_amdgcn_wmma_f32_16x16x32_bf16(
        /*neg_a=*/false, a.bf, /*neg_b=*/false, b.bf,
        /*c_mod=*/(short)0, c, /*reuse_a=*/false, /*reuse_b=*/false);
}

// LDS byte offset of a __shared__ object: flat LDS addresses carry the LDS
// offset in addr[31:0] (ISA aperture mapping), so truncation is exact.
__device__ __forceinline__ unsigned lds_off32(const void* p) {
    return (unsigned)(unsigned long long)(uintptr_t)p;
}

// Async global->LDS 16-byte copy (GVS mode: saddr base + 32-bit voffset).
// Tracked by ASYNCcnt; completion via s_wait_asynccnt.
__device__ __forceinline__ void async_ld16(unsigned lds_off, unsigned g_off,
                                           unsigned long long base) {
    asm volatile("global_load_async_to_lds_b128 %0, %1, %2"
                 :: "v"(lds_off), "v"(g_off), "s"(base) : "memory");
}

__device__ __forceinline__ void wait_async0() {
    asm volatile("s_wait_asynccnt 0x0" ::: "memory");
}

// ---------------------------------------------------------------------------
// Stage 0: fp32 -> bf16 conversion (RNE), 4 elements per thread
// ---------------------------------------------------------------------------
__global__ __launch_bounds__(256)
void cvt_f32_bf16(const float* __restrict__ src, unsigned short* __restrict__ dst, int n) {
    int i = (blockIdx.x * blockDim.x + threadIdx.x) * 4;
    if (i >= n) return;
    float4 f = *(const float4*)(src + i);
    unsigned long long pk =
          (unsigned long long)f32_to_bf16_rne(f.x)
        | ((unsigned long long)f32_to_bf16_rne(f.y) << 16)
        | ((unsigned long long)f32_to_bf16_rne(f.z) << 32)
        | ((unsigned long long)f32_to_bf16_rne(f.w) << 48);
    *(unsigned long long*)(dst + i) = pk;
}

// ---------------------------------------------------------------------------
// Stage 1: QKV projections  y = x @ W^T + b   (bf16 WMMA, f32 accumulate)
// One wave computes a 16(M) x 64(N) output strip of one of {Q,K,V}.
// Q scaled by 1/sqrt(DH). Q,K stored [B,H,S,DH]; V stored transposed [B,H,DH,S].
// x and all three weight matrices are L2-resident (14 MB bf16 total), so the
// per-tile re-reads hit L2; the loop runs at WMMA rate.
// ---------------------------------------------------------------------------
__global__ __launch_bounds__(256)
void qkv_wmma(const unsigned short* __restrict__ xb,   // [B*S, E] bf16
              const unsigned short* __restrict__ wq,   // [E, E] bf16 (row = out feature)
              const unsigned short* __restrict__ wk,
              const unsigned short* __restrict__ wv,
              const float* __restrict__ bq,
              const float* __restrict__ bk,
              const float* __restrict__ bv,
              unsigned short* __restrict__ qd,          // [B,H,S,DH]
              unsigned short* __restrict__ kd,          // [B,H,S,DH]
              unsigned short* __restrict__ vtd)         // [B,H,DH,S]
{
    const int lane = threadIdx.x & 31;
    const int wave = threadIdx.x >> 5;
    const int job  = blockIdx.x * 8 + wave;   // 3 * 256 * 16 = 12288 jobs
    const int mat  = job >> 12;               // 0:Q 1:K 2:V
    const int rem  = job & 4095;
    const int mt   = rem >> 4;                // 256 M-tiles over B*S=4096 rows
    const int ns   = rem & 15;                // 16 N-strips of 64 (== head index)
    const int row0 = mt * 16;
    const int n0   = ns * 64;

    const unsigned short* W   = (mat == 0) ? wq : (mat == 1) ? wk : wv;
    const float*          bia = (mat == 0) ? bq : (mat == 1) ? bk : bv;

    const int ml = lane & 15;        // A: row / B: column / C: N index
    const int hi = lane >> 4;        // lane-half select
    const int hs = hi * 8;           // unused (A and B frags both use ks)
    const int ks = hi * 16;          // B-frag K sub-offset

    const v8f vzero = {0.f, 0.f, 0.f, 0.f, 0.f, 0.f, 0.f, 0.f};
    v8f acc[4] = {vzero, vzero, vzero, vzero};

    const unsigned short* arow  = xb + (row0 + ml) * EE;
    const unsigned short* brow0 = W + (n0 +  0 + ml) * EE + ks;
    const unsigned short* brow1 = W + (n0 + 16 + ml) * EE + ks;
    const unsigned short* brow2 = W + (n0 + 32 + ml) * EE + ks;
    const unsigned short* brow3 = W + (n0 + 48 + ml) * EE + ks;

    for (int k0 = 0; k0 < EE; k0 += 32) {
        __builtin_prefetch(arow + k0 + 128, 0, 0);   // global_prefetch_b8
        Frag a;
        a.u[0] = *(const v4u*)(arow + k0 + ks);       // K = ks..ks+7
        a.u[1] = *(const v4u*)(arow + k0 + ks + 8);   // K = ks+8..+15
        Frag b;
        b.u[0] = *(const v4u*)(brow0 + k0);
        b.u[1] = *(const v4u*)(brow0 + k0 + 8);
        acc[0] = wmma_bf16(a, b, acc[0]);
        b.u[0] = *(const v4u*)(brow1 + k0);
        b.u[1] = *(const v4u*)(brow1 + k0 + 8);
        acc[1] = wmma_bf16(a, b, acc[1]);
        b.u[0] = *(const v4u*)(brow2 + k0);
        b.u[1] = *(const v4u*)(brow2 + k0 + 8);
        acc[2] = wmma_bf16(a, b, acc[2]);
        b.u[0] = *(const v4u*)(brow3 + k0);
        b.u[1] = *(const v4u*)(brow3 + k0 + 8);
        acc[3] = wmma_bf16(a, b, acc[3]);
    }

    const float qscale = 0.125f;  // 1/sqrt(64)
#pragma unroll
    for (int t = 0; t < 4; ++t) {
        float bval = bia[n0 + t * 16 + ml];
#pragma unroll
        for (int r = 0; r < 8; ++r) {
            int g = row0 + r + 8 * hi;     // global row in [0, B*S)
            int bidx = g >> 11;            // batch
            int s    = g & (SS - 1);       // seq pos
            int d    = t * 16 + ml;        // dim within head (strip == head ns)
            float val = acc[t][r] + bval;
            if (mat == 0) {
                qd[((bidx * HH + ns) * SS + s) * DH + d] = f32_to_bf16_rne(val * qscale);
            } else if (mat == 1) {
                kd[((bidx * HH + ns) * SS + s) * DH + d] = f32_to_bf16_rne(val);
            } else {
                vtd[((bidx * HH + ns) * DH + d) * SS + s] = f32_to_bf16_rne(val);
            }
        }
    }
}

// ---------------------------------------------------------------------------
// Stage 2: flash attention, cooperative within a workgroup.
// One block (8 waves) owns one (b,h) and 128 queries (16 per wave).
// Each 32-key chunk of K [32,64] and V^T [64,32] (4 KB each) is staged into
// LDS once per block with GLOBAL_LOAD_ASYNC_TO_LDS_B128 (double-buffered,
// overlapped with compute), then all 8 waves consume it via ds_load_b128
// B-fragments -> 8x less L2 traffic for K/V than per-wave streaming.
// Probs are repacked C->A layout through a per-wave LDS tile.
// Output fp32 in [B,H,S,DH]-flat order (== reference's quirky reshape).
// ---------------------------------------------------------------------------
__global__ __launch_bounds__(256)
void attn_wmma(const unsigned short* __restrict__ qd,
               const unsigned short* __restrict__ kd,
               const unsigned short* __restrict__ vtd,
               float* __restrict__ out)
{
    __shared__ unsigned short kbuf[2][32 * 64];   // [key][dim], 4 KB per buffer
    __shared__ unsigned short vbuf[2][64 * 32];   // [dim][key], 4 KB per buffer
    __shared__ unsigned short plds[8][16 * 32];   // per-wave probs tile, 1 KB

    const int tid  = threadIdx.x;
    const int lane = tid & 31;
    const int wave = tid >> 5;
    const int bh   = blockIdx.x >> 4;          // (b*H + h), 32 values
    const int qb   = blockIdx.x & 15;          // query block of 128
    const int m0   = qb * 128 + wave * 16;

    const unsigned short* qh = qd  + (size_t)bh * SS * DH;
    const unsigned short* kh = kd  + (size_t)bh * SS * DH;
    const unsigned short* vh = vtd + (size_t)bh * DH * SS;

    // ---- staging geometry: 256 threads x 16 B cover each 4 KB chunk ----
    const unsigned long long kbase = (unsigned long long)(uintptr_t)kh;
    const unsigned long long vbase = (unsigned long long)(uintptr_t)vh;
    const unsigned kg_lane = (unsigned)tid * 16u;            // byte off in K chunk
    const int      vd_row  = tid >> 2;                       // dim row 0..63
    const int      vq_seg  = tid & 3;                        // 16B segment in row
    const unsigned vg_lane = (unsigned)vd_row * (SS * 2u) + (unsigned)vq_seg * 16u;
    const unsigned klds[2] = { lds_off32(&kbuf[0][0]) + kg_lane,
                               lds_off32(&kbuf[1][0]) + kg_lane };
    const unsigned vlds[2] = { lds_off32(&vbuf[0][0]) + (unsigned)(vd_row * 64 + vq_seg * 16),
                               lds_off32(&vbuf[1][0]) + (unsigned)(vd_row * 64 + vq_seg * 16) };

    const int ml = lane & 15;
    const int hi = lane >> 4;
    const int hs = hi * 8;
    const int ks = hi * 16;

    // Q A-fragments for k-steps 0 and 32 (K = DH = 64), loaded once.
    Frag aq0, aq1;
    {
        const unsigned short* qrow = qh + (m0 + ml) * DH;
        aq0.u[0] = *(const v4u*)(qrow + ks);
        aq0.u[1] = *(const v4u*)(qrow + ks + 8);
        aq1.u[0] = *(const v4u*)(qrow + 32 + ks);
        aq1.u[1] = *(const v4u*)(qrow + 40 + ks);
    }

    const v8f vzero = {0.f, 0.f, 0.f, 0.f, 0.f, 0.f, 0.f, 0.f};
    v8f accv[4] = {vzero, vzero, vzero, vzero};   // 16 x 64 f32 output accumulator
    float mx[8], lsum[8];
#pragma unroll
    for (int r = 0; r < 8; ++r) { mx[r] = -1e30f; lsum[r] = 0.f; }

    unsigned short* lbase = &plds[wave][0];

    // ---- prologue: stage chunk 0 into buffer 0 ----
    async_ld16(klds[0], kg_lane, kbase);                    // K bytes [0, 4096)
    async_ld16(vlds[0], vg_lane, vbase);                    // V^T segs at sk=0
    wait_async0();
    __syncthreads();

    for (int sk = 0; sk < SS; sk += 32) {
        const int cur = (sk >> 5) & 1;

        // ---- issue async stage of next chunk into the other buffer ----
        // (that buffer's readers finished before last iteration's barrier)
        if (sk + 32 < SS) {
            const unsigned sk2 = (unsigned)(sk + 32);
            async_ld16(klds[cur ^ 1], sk2 * 128u + kg_lane, kbase);
            async_ld16(vlds[cur ^ 1], sk2 * 2u + vg_lane, vbase);
        }

        const unsigned short* kb = &kbuf[cur][0];
        const unsigned short* vb = &vbuf[cur][0];

        // ---- scores: two 16q x 16k tiles, K = DH = 64 (2 bf16 k-steps) ----
        v8f c0 = vzero, c1 = vzero;
        {
            const unsigned short* k0row = kb + ml * DH;          // keys sk..sk+15
            const unsigned short* k1row = kb + (16 + ml) * DH;   // keys sk+16..sk+31
            Frag b;
            b.u[0] = *(const v4u*)(k0row + ks);
            b.u[1] = *(const v4u*)(k0row + ks + 8);
            c0 = wmma_bf16(aq0, b, c0);
            b.u[0] = *(const v4u*)(k0row + 32 + ks);
            b.u[1] = *(const v4u*)(k0row + 40 + ks);
            c0 = wmma_bf16(aq1, b, c0);
            b.u[0] = *(const v4u*)(k1row + ks);
            b.u[1] = *(const v4u*)(k1row + ks + 8);
            c1 = wmma_bf16(aq0, b, c1);
            b.u[0] = *(const v4u*)(k1row + 32 + ks);
            b.u[1] = *(const v4u*)(k1row + 40 + ks);
            c1 = wmma_bf16(aq1, b, c1);
        }

        // ---- online softmax per accumulator row (row M spread over 16 lanes) ----
#pragma unroll
        for (int r = 0; r < 8; ++r) {
            float s0 = c0[r], s1 = c1[r];
            float t = fmaxf(s0, s1);
            t = fmaxf(t, __shfl_xor(t, 1, 32));
            t = fmaxf(t, __shfl_xor(t, 2, 32));
            t = fmaxf(t, __shfl_xor(t, 4, 32));
            t = fmaxf(t, __shfl_xor(t, 8, 32));
            float mnew = fmaxf(mx[r], t);
            float p0 = __expf(s0 - mnew);
            float p1 = __expf(s1 - mnew);
            float fac = __expf(mx[r] - mnew);
            mx[r] = mnew;
            float rs = p0 + p1;
            rs += __shfl_xor(rs, 1, 32);
            rs += __shfl_xor(rs, 2, 32);
            rs += __shfl_xor(rs, 4, 32);
            rs += __shfl_xor(rs, 8, 32);
            lsum[r] = lsum[r] * fac + rs;
#pragma unroll
            for (int t4 = 0; t4 < 4; ++t4) accv[t4][r] *= fac;
            int m = r + 8 * hi;                         // C-layout row
            lbase[m * 32 + ml]      = f32_to_bf16_rne(p0);
            lbase[m * 32 + 16 + ml] = f32_to_bf16_rne(p1);
        }

        // LDS ops are in-order per wave; fence the compiler + counters.
        asm volatile("s_wait_dscnt 0" ::: "memory");

        // ---- repack probs into A-layout (16 x 32, K = 32 keys) ----
        Frag ap;
        ap.u[0] = *(const v4u*)(lbase + ml * 32 + ks);
        ap.u[1] = *(const v4u*)(lbase + ml * 32 + ks + 8);

        // ---- PV: out += probs @ V   (B from V^T tile in LDS) ----
#pragma unroll
        for (int t4 = 0; t4 < 4; ++t4) {
            const unsigned short* vrow = vb + (t4 * 16 + ml) * 32;
            Frag bv;
            bv.u[0] = *(const v4u*)(vrow + ks);
            bv.u[1] = *(const v4u*)(vrow + ks + 8);
            accv[t4] = wmma_bf16(ap, bv, accv[t4]);
        }

        // ---- publish next chunk, retire reads of current buffers ----
        wait_async0();        // own async loads of chunk sk+32 complete
        __syncthreads();      // all waves: loads landed, reads of `cur` done
    }

    // ---- normalize and store ----
    float* obase = out + ((size_t)bh * SS + m0) * DH;
#pragma unroll
    for (int r = 0; r < 8; ++r) {
        float inv = 1.0f / lsum[r];
        int m = r + 8 * hi;
#pragma unroll
        for (int t4 = 0; t4 < 4; ++t4) {
            obase[m * DH + t4 * 16 + ml] = accv[t4][r] * inv;
        }
    }
}

// ---------------------------------------------------------------------------
// Host entry
// ---------------------------------------------------------------------------
extern "C" void kernel_launch(void* const* d_in, const int* in_sizes, int n_in,
                              void* d_out, int out_size, void* d_ws, size_t ws_size,
                              hipStream_t stream) {
    (void)in_sizes; (void)n_in; (void)out_size; (void)ws_size;

    const float* x  = (const float*)d_in[0];
    const float* Wq = (const float*)d_in[1];
    const float* bq = (const float*)d_in[2];
    const float* Wk = (const float*)d_in[3];
    const float* bk = (const float*)d_in[4];
    const float* Wv = (const float*)d_in[5];
    const float* bv = (const float*)d_in[6];
    float* out = (float*)d_out;

    char* ws = (char*)d_ws;
    unsigned short* xb  = (unsigned short*)ws; ws += (size_t)BB * SS * EE * 2;  // 8 MB
    unsigned short* wqb = (unsigned short*)ws; ws += (size_t)EE * EE * 2;       // 2 MB
    unsigned short* wkb = (unsigned short*)ws; ws += (size_t)EE * EE * 2;
    unsigned short* wvb = (unsigned short*)ws; ws += (size_t)EE * EE * 2;
    unsigned short* qd  = (unsigned short*)ws; ws += (size_t)BB * HH * SS * DH * 2; // 8 MB
    unsigned short* kd  = (unsigned short*)ws; ws += (size_t)BB * HH * SS * DH * 2;
    unsigned short* vtd = (unsigned short*)ws; ws += (size_t)BB * HH * DH * SS * 2;

    const int nx = BB * SS * EE;    // 4,194,304
    const int nw = EE * EE;         // 1,048,576
    cvt_f32_bf16<<<nx / 4 / 256, 256, 0, stream>>>(x,  xb,  nx);
    cvt_f32_bf16<<<nw / 4 / 256, 256, 0, stream>>>(Wq, wqb, nw);
    cvt_f32_bf16<<<nw / 4 / 256, 256, 0, stream>>>(Wk, wkb, nw);
    cvt_f32_bf16<<<nw / 4 / 256, 256, 0, stream>>>(Wv, wvb, nw);

    // 3 mats * 256 M-tiles * 16 N-strips = 12288 wave jobs / 8 waves = 1536 blocks
    qkv_wmma<<<1536, 256, 0, stream>>>(xb, wqb, wkb, wvb, bq, bk, bv, qd, kd, vtd);

    // 32 (b,h) * 16 query-blocks of 128 = 512 blocks, 8 waves each
    attn_wmma<<<512, 256, 0, stream>>>(qd, kd, vtd, out);
}
